// LoLGNN_68341519614847
// MI455X (gfx1250) — matrix operand order repaired and checked by reading, not compiled, not run on hardware
//
#include <hip/hip_runtime.h>

typedef _Float16 v8h  __attribute__((ext_vector_type(8)));
typedef _Float16 v16h __attribute__((ext_vector_type(16)));
typedef float    v8f  __attribute__((ext_vector_type(8)));

#define HDIM 128

// ---------------------------------------------------------------- fill zeros
__global__ void fill_zero_f32(float* p, long long n) {
    long long i = (long long)blockIdx.x * blockDim.x + threadIdx.x;
    long long stride = (long long)gridDim.x * blockDim.x;
    for (; i < n; i += stride) p[i] = 0.0f;
}

// ---------------------------------------------------------------- weight prep
// wl_h[(l*3+m)*16384+j] = half(Wl), wr_h[l*16384+j] = half(sum_m Wr),
// bsum[l*128+j] = sum_m b
__global__ void prep_weights(const float* __restrict__ Wl,
                             const float* __restrict__ Wr,
                             const float* __restrict__ b,
                             _Float16* __restrict__ wl_h,
                             _Float16* __restrict__ wr_h,
                             float* __restrict__ bsum) {
    int i = blockIdx.x * blockDim.x + threadIdx.x;
    if (i >= 2 * 16384) return;
    int l = i >> 14, j = i & 16383;
    float s = 0.0f;
#pragma unroll
    for (int m = 0; m < 3; ++m) {
        float w = Wl[(size_t)(l * 3 + m) * 16384 + j];
        wl_h[(size_t)(l * 3 + m) * 16384 + j] = (_Float16)w;
        s += Wr[(size_t)(l * 3 + m) * 16384 + j];
    }
    wr_h[i] = (_Float16)s;
    if (j < HDIM) {
        float bs = 0.0f;
#pragma unroll
        for (int m = 0; m < 3; ++m) bs += b[(l * 3 + m) * HDIM + j];
        bsum[l * HDIM + j] = bs;
    }
}

// ---------------------------------------------------------------- encoders
// player: feat85 = [x(0,6,7,8,9), emb0..emb4(16 each)]; p = Wp(128x85)@feat + bp
__global__ __launch_bounds__(128) void encode_player(
    const float* __restrict__ xp, const float* __restrict__ emb,
    const float* __restrict__ Wp, const float* __restrict__ bp,
    float* __restrict__ out, int N, int chunk) {
    __shared__ float sW[HDIM * 85];
    __shared__ float sF[85];
    int t = threadIdx.x;
    for (int i = t; i < HDIM * 85; i += 128) sW[i] = Wp[i];
    __syncthreads();
    int r0 = blockIdx.x * chunk;
    int r1 = min(r0 + chunk, N);
    for (int r = r0; r < r1; ++r) {
        if (t < 85) {
            float v;
            if (t < 5) {
                const int cmap[5] = {0, 6, 7, 8, 9};
                v = xp[(size_t)r * 10 + cmap[t]];
            } else {
                int q = t - 5;
                int i = q >> 4, j = q & 15;
                int idx = (int)xp[(size_t)r * 10 + i + 1];
                idx = min(max(idx, 0), 199);
                v = emb[(size_t)(i * 200 + idx) * 16 + j];
            }
            sF[t] = v;
        }
        __syncthreads();
        float acc = bp[t];
#pragma unroll 5
        for (int k = 0; k < 85; ++k) acc += sW[t * 85 + k] * sF[k];
        out[(size_t)r * HDIM + t] = acc;
        __syncthreads();
    }
}

// history: feat26 = [x(1,2,4,5,6,7), emb_h0(16), emb_h3(4)]
__global__ __launch_bounds__(128) void encode_hist(
    const float* __restrict__ xh, const float* __restrict__ e0,
    const float* __restrict__ e3, const float* __restrict__ Wh,
    const float* __restrict__ bh, float* __restrict__ out, int N, int chunk) {
    __shared__ float sW[HDIM * 26];
    __shared__ float sF[26];
    int t = threadIdx.x;
    for (int i = t; i < HDIM * 26; i += 128) sW[i] = Wh[i];
    __syncthreads();
    int r0 = blockIdx.x * chunk;
    int r1 = min(r0 + chunk, N);
    for (int r = r0; r < r1; ++r) {
        if (t < 26) {
            float v;
            if (t < 6) {
                const int cmap[6] = {1, 2, 4, 5, 6, 7};
                v = xh[(size_t)r * 8 + cmap[t]];
            } else if (t < 22) {
                int idx = (int)xh[(size_t)r * 8 + 0];
                idx = min(max(idx, 0), 1999);
                v = e0[(size_t)idx * 16 + (t - 6)];
            } else {
                int idx = (int)xh[(size_t)r * 8 + 3];
                idx = min(max(idx, 0), 9);
                v = e3[idx * 4 + (t - 22)];
            }
            sF[t] = v;
        }
        __syncthreads();
        float acc = bh[t];
#pragma unroll 2
        for (int k = 0; k < 26; ++k) acc += sW[t * 26 + k] * sF[k];
        out[(size_t)r * HDIM + t] = acc;
        __syncthreads();
    }
}

// ---------------------------------------------------------------- graph ops
__global__ void count_edges(const int* __restrict__ dst, int E,
                            float* __restrict__ cnt) {
    int i = blockIdx.x * blockDim.x + threadIdx.x;
    int stride = gridDim.x * blockDim.x;
    for (; i < E; i += stride) atomicAdd(&cnt[dst[i]], 1.0f);
}

// one wave32 per edge; lane handles 4 contiguous features (float4)
__global__ __launch_bounds__(256) void scatter_add(
    const float* __restrict__ src_feat, const int* __restrict__ src,
    const int* __restrict__ dst, int E, float* __restrict__ acc) {
    int wv = (int)((blockIdx.x * (unsigned)blockDim.x + threadIdx.x) >> 5);
    int lane = threadIdx.x & 31;
    if (wv >= E) return;
    int s = src[wv], d = dst[wv];
    const float4 v = *(const float4*)(src_feat + (size_t)s * HDIM + lane * 4);
    float* a = acc + (size_t)d * HDIM + lane * 4;
    atomicAdd(a + 0, v.x);
    atomicAdd(a + 1, v.y);
    atomicAdd(a + 2, v.z);
    atomicAdd(a + 3, v.w);
}

// out_h[i] = half( acc[i] / max(cnt[row],1) )   (cnt==null -> plain convert)
__global__ void scale_to_half(const float* __restrict__ acc,
                              const float* __restrict__ cnt,
                              _Float16* __restrict__ out, long long n) {
    long long i = (long long)blockIdx.x * blockDim.x + threadIdx.x;
    long long stride = (long long)gridDim.x * blockDim.x;
    for (; i < n; i += stride) {
        float v = acc[i];
        if (cnt) v /= fmaxf(cnt[i >> 7], 1.0f);
        out[i] = (_Float16)v;
    }
}

// ---------------------------------------------------------------- WMMA GEMM
// out = relu(A0@W0^T + A1@W1^T + A2@W2^T + A3@W3^T + bias), N=K=128.
// 8 waves/block; wave w -> 16x16 tile (rows 16*bid, cols 16*w).
__device__ inline v16h load_a16(const _Float16* __restrict__ A, int row, int kb) {
    v8h lo = *(const v8h*)(A + (size_t)row * HDIM + kb);
    v8h hi = *(const v8h*)(A + (size_t)row * HDIM + kb + 16);
    return __builtin_shufflevector(lo, hi, 0, 1, 2, 3, 4, 5, 6, 7,
                                   8, 9, 10, 11, 12, 13, 14, 15);
}

__global__ __launch_bounds__(256) void sage_gemm_relu(
    const _Float16* __restrict__ A0, const _Float16* __restrict__ A1,
    const _Float16* __restrict__ A2, const _Float16* __restrict__ A3,
    const _Float16* __restrict__ W0, const _Float16* __restrict__ W1,
    const _Float16* __restrict__ W2, const _Float16* __restrict__ W3,
    const float* __restrict__ bias, float* __restrict__ outF,
    _Float16* __restrict__ outH) {
    int wave = threadIdx.x >> 5;
    int lane = threadIdx.x & 31;
    int m0 = blockIdx.x * 16;
    int n0 = wave * 16;
    int arow = m0 + (lane & 15);          // A: row per lane
    int ka = (lane >> 4) * 8;             // A: K sub-base (halves 0-7 / 8-15 +16)
    int wrow = n0 + (lane & 15);          // B: column n == W row n
    int kb = (lane >> 4) * 8;             // B: K sub-base (halves 0-7 / 8-15 +16)

    v8f acc = {};
    const _Float16* As[4] = {A0, A1, A2, A3};
    const _Float16* Ws[4] = {W0, W1, W2, W3};
#pragma unroll
    for (int mtx = 0; mtx < 4; ++mtx) {
        const _Float16* A = As[mtx];
        const _Float16* W = Ws[mtx];
#pragma unroll
        for (int kc = 0; kc < 4; ++kc) {
            int k0 = kc * 32;
            v16h a = load_a16(A, arow, k0 + ka);
            v16h b = load_a16(W, wrow, k0 + kb);
            acc = __builtin_amdgcn_wmma_f32_16x16x32_f16(
                false, a, false, b, (short)0, acc, false, false);
        }
    }
    // D layout: lane -> n = n0+(lane&15); VGPR j -> m = m0 + j + (lane>=16 ? 8:0)
    float bval = bias[n0 + (lane & 15)];
    int mbase = m0 + ((lane >> 4) << 3);
    int n = n0 + (lane & 15);
#pragma unroll
    for (int j = 0; j < 8; ++j) {
        float v = acc[j] + bval;
        v = v > 0.0f ? v : 0.0f;
        size_t off = (size_t)(mbase + j) * HDIM + n;
        outF[off] = v;
        outH[off] = (_Float16)v;
    }
}

// ---------------------------------------------------------------- pooling
// batch = arange//10 -> contiguous groups of 10. out[g] = mean(p[g])@Wc + bc
__global__ __launch_bounds__(256) void pool_readout(
    const float* __restrict__ p, const float* __restrict__ Wc,
    const float* __restrict__ bc, float* __restrict__ out, int G) {
    int wv = (int)((blockIdx.x * (unsigned)blockDim.x + threadIdx.x) >> 5);
    int lane = threadIdx.x & 31;
    if (wv >= G) return;
    float s = 0.0f;
#pragma unroll
    for (int c = lane; c < HDIM; c += 32) {
        float colsum = 0.0f;
#pragma unroll
        for (int r = 0; r < 10; ++r)
            colsum += p[(size_t)(wv * 10 + r) * HDIM + c];
        s += colsum * Wc[c];
    }
#pragma unroll
    for (int off = 16; off; off >>= 1) s += __shfl_xor(s, off, 32);
    if (lane == 0) out[wv] = s * 0.1f + bc[0];
}

// ---------------------------------------------------------------- launcher
extern "C" void kernel_launch(void* const* d_in, const int* in_sizes, int n_in,
                              void* d_out, int out_size, void* d_ws, size_t ws_size,
                              hipStream_t stream) {
    const float* x_player = (const float*)d_in[0];
    const float* x_history = (const float*)d_in[1];
    const int* e_team = (const int*)d_in[2];
    const int* e_enemy = (const int*)d_in[3];
    const int* e_hist = (const int*)d_in[4];
    const float* emb_player = (const float*)d_in[6];
    const float* emb_h0 = (const float*)d_in[7];
    const float* emb_h3 = (const float*)d_in[8];
    const float* Wp = (const float*)d_in[9];
    const float* bp = (const float*)d_in[10];
    const float* Wh = (const float*)d_in[11];
    const float* bh = (const float*)d_in[12];
    const float* sage_Wl = (const float*)d_in[13];
    const float* sage_b = (const float*)d_in[14];
    const float* sage_Wr = (const float*)d_in[15];
    const float* Wc = (const float*)d_in[16];
    const float* bc = (const float*)d_in[17];
    float* out = (float*)d_out;

    const int NP = in_sizes[0] / 10;         // 200000
    const int NH = in_sizes[1] / 8;          // 500000
    const int NG = out_size;                 // 20000
    const int ET = in_sizes[2] / 2;          // 800000
    const int EE = in_sizes[3] / 2;          // 1000000
    const int EH = in_sizes[4] / 2;          // 500000
    const long long NPF = (long long)NP * HDIM;

    // workspace carve-up
    char* base = (char*)d_ws;
    size_t off = 0;
    auto alloc = [&](size_t bytes) -> char* {
        off = (off + 255) & ~(size_t)255;
        char* p = base + off;
        off += bytes;
        return p;
    };
    float* pA   = (float*)alloc(NPF * 4);
    float* pB   = (float*)alloc(NPF * 4);
    float* hist = (float*)alloc((size_t)NH * HDIM * 4);
    float* acc0 = (float*)alloc(NPF * 4);
    float* acc1 = (float*)alloc(NPF * 4);
    float* acc2 = (float*)alloc(NPF * 4);
    float* cnt0 = (float*)alloc((size_t)NP * 4);
    float* cnt1 = (float*)alloc((size_t)NP * 4);
    float* cnt2 = (float*)alloc((size_t)NP * 4);
    _Float16* aggT_h = (_Float16*)alloc(NPF * 2);
    _Float16* aggE_h = (_Float16*)alloc(NPF * 2);
    _Float16* aggH_h = (_Float16*)alloc(NPF * 2);
    _Float16* pA_h   = (_Float16*)alloc(NPF * 2);
    _Float16* pB_h   = (_Float16*)alloc(NPF * 2);
    _Float16* wl_h   = (_Float16*)alloc((size_t)6 * 16384 * 2);
    _Float16* wr_h   = (_Float16*)alloc((size_t)2 * 16384 * 2);
    float* bsum      = (float*)alloc((size_t)2 * HDIM * 4);
    (void)ws_size; (void)n_in;

    const int FB = 1024;  // fill grid cap
    auto fill = [&](float* p, long long n) {
        int blocks = (int)min((long long)FB * 64, (n + 255) / 256);
        fill_zero_f32<<<blocks, 256, 0, stream>>>(p, n);
    };

    // 0) weights
    prep_weights<<<(2 * 16384 + 255) / 256, 256, 0, stream>>>(
        sage_Wl, sage_Wr, sage_b, wl_h, wr_h, bsum);

    // 1) encoders
    encode_player<<<(NP + 31) / 32, 128, 0, stream>>>(
        x_player, emb_player, Wp, bp, pA, NP, 32);
    encode_hist<<<(NH + 31) / 32, 128, 0, stream>>>(
        x_history, emb_h0, emb_h3, Wh, bh, hist, NH, 32);

    // 2) zero accumulators + counts
    fill(acc0, NPF); fill(acc1, NPF); fill(acc2, NPF);
    fill(cnt0, NP);  fill(cnt1, NP);  fill(cnt2, NP);

    // 3) edge counts (layer-invariant)
    count_edges<<<(ET + 255) / 256, 256, 0, stream>>>(e_team + ET, ET, cnt0);
    count_edges<<<(EE + 255) / 256, 256, 0, stream>>>(e_enemy + EE, EE, cnt1);
    count_edges<<<(EH + 255) / 256, 256, 0, stream>>>(e_hist + EH, EH, cnt2);

    // 4) hist aggregation (layer-invariant: hist never updates)
    scatter_add<<<(EH * 32 + 255) / 256, 256, 0, stream>>>(
        hist, e_hist, e_hist + EH, EH, acc2);
    {
        int blocks = (int)min((long long)65536, (NPF + 255) / 256);
        scale_to_half<<<blocks, 256, 0, stream>>>(acc2, cnt2, aggH_h, NPF);
    }

    int scale_blocks = (int)min((long long)65536, (NPF + 255) / 256);

    // ---- layer 1 ----
    scatter_add<<<(ET * 32 + 255) / 256, 256, 0, stream>>>(
        pA, e_team, e_team + ET, ET, acc0);
    scatter_add<<<(EE * 32 + 255) / 256, 256, 0, stream>>>(
        pA, e_enemy, e_enemy + EE, EE, acc1);
    scale_to_half<<<scale_blocks, 256, 0, stream>>>(acc0, cnt0, aggT_h, NPF);
    scale_to_half<<<scale_blocks, 256, 0, stream>>>(acc1, cnt1, aggE_h, NPF);
    scale_to_half<<<scale_blocks, 256, 0, stream>>>(pA, nullptr, pA_h, NPF);
    sage_gemm_relu<<<NP / 16, 256, 0, stream>>>(
        aggT_h, aggE_h, aggH_h, pA_h,
        wl_h + 0 * 16384, wl_h + 1 * 16384, wl_h + 2 * 16384, wr_h,
        bsum, pB, pB_h);

    // ---- layer 2 ----
    fill(acc0, NPF); fill(acc1, NPF);
    scatter_add<<<(ET * 32 + 255) / 256, 256, 0, stream>>>(
        pB, e_team, e_team + ET, ET, acc0);
    scatter_add<<<(EE * 32 + 255) / 256, 256, 0, stream>>>(
        pB, e_enemy, e_enemy + EE, EE, acc1);
    scale_to_half<<<scale_blocks, 256, 0, stream>>>(acc0, cnt0, aggT_h, NPF);
    scale_to_half<<<scale_blocks, 256, 0, stream>>>(acc1, cnt1, aggE_h, NPF);
    sage_gemm_relu<<<NP / 16, 256, 0, stream>>>(
        aggT_h, aggE_h, aggH_h, pB_h,
        wl_h + 3 * 16384, wl_h + 4 * 16384, wl_h + 5 * 16384, wr_h + 16384,
        bsum + HDIM, pA, pA_h);

    // 5) readout
    pool_readout<<<(NG * 32 + 255) / 256, 256, 0, stream>>>(pA, Wc, bc, out, NG);
}
